// GeneInteractionDango_83270825935262
// MI455X (gfx1250) — compile-verified, matrix operand order checked by reading, not run on hardware
//
#include <hip/hip_runtime.h>
#include <hip/hip_bf16.h>

typedef __attribute__((ext_vector_type(16))) _Float16 v16h;
typedef __attribute__((ext_vector_type(8)))  _Float16 v8h;
typedef __attribute__((ext_vector_type(4)))  _Float16 v4h;
typedef __attribute__((ext_vector_type(2)))  _Float16 v2h;
typedef __attribute__((ext_vector_type(8)))  float    v8f;

#define H        128
#define RPB      48          // rows per block  (16 groups * 3)
#define GPB      16          // groups per block
#define NTHREADS 256         // 8 waves (wave32)

#define WMMA_F16(a, b, c) \
  __builtin_amdgcn_wmma_f32_16x16x32_f16(false, (a), false, (b), (short)0, (c), false, false)

// ---------------------------------------------------------------------------
// Pre-kernel: convert the 9 fp32 128x128 weight matrices into f16 B-fragment
// layout: [matrix][col_tile(8)][k_step(4)][lane(32)][elem(16)] so a WMMA B
// fragment is one contiguous 32B load per lane.
// matrix order: 0=Ws, 1..4 = layer0 {q,k,v,o}, 5..8 = layer1 {q,k,v,o}
// ---------------------------------------------------------------------------
__global__ void repack_weights(const float* __restrict__ Ws,
                               const float* __restrict__ Wq,
                               const float* __restrict__ Wk,
                               const float* __restrict__ Wv,
                               const float* __restrict__ Wo,
                               _Float16* __restrict__ out) {
  int idx = blockIdx.x * blockDim.x + threadIdx.x;
  if (idx >= 9 * 16384) return;
  int m    = idx >> 14;
  int r    = idx & 16383;
  int e    = r & 15;
  int lane = (r >> 4) & 31;
  int ks   = (r >> 9) & 3;
  int ct   = (r >> 11) & 7;
  // 16-bit fragment K index pattern (mirrors documented A layout):
  int k = ks * 32 + ((e & 8) * 2 + ((lane >> 4) << 3) + (e & 7));
  int n = (ct << 4) + (lane & 15);
  const float* src;
  if (m == 0) {
    src = Ws;
  } else {
    int l = (m - 1) >> 2;
    int w = (m - 1) & 3;
    const float* base = (w == 0) ? Wq : (w == 1) ? Wk : (w == 2) ? Wv : Wo;
    src = base + l * 16384;
  }
  out[idx] = (_Float16)src[k * H + n];
}

// ---------------------------------------------------------------------------
// A-fragment gather from row-major f16 LDS tile (two contiguous 16B loads).
// ---------------------------------------------------------------------------
__device__ __forceinline__ v16h load_afrag(const _Float16* xa, int m, int ks,
                                           int khalf8) {
  const _Float16* row = xa + m * H + ks * 32 + khalf8;
  v8h lo = *(const v8h*)(row);
  v8h hi = *(const v8h*)(row + 16);
  v16h a;
#pragma unroll
  for (int i = 0; i < 8; ++i) { a[i] = lo[i]; a[8 + i] = hi[i]; }
  return a;
}

// ---------------------------------------------------------------------------
// Generic workgroup GEMM (static proj, output proj).
// MODE 0: relu(c+b)   MODE 2: out += beta*(c+b)
// ---------------------------------------------------------------------------
template <int MODE>
__device__ __forceinline__ void wg_gemm(const _Float16* __restrict__ wfrag,
                                        const float* __restrict__ bias,
                                        const _Float16* __restrict__ xa,
                                        float* __restrict__ out, float beta) {
  const int wave   = threadIdx.x >> 5;
  const int lane   = threadIdx.x & 31;
  const int khalf8 = (lane >> 4) << 3;
  const _Float16* wb = wfrag + wave * 2048 + lane * 16;
  const v16h b0 = *(const v16h*)(wb);
  const v16h b1 = *(const v16h*)(wb + 512);
  const v16h b2 = *(const v16h*)(wb + 1024);
  const v16h b3 = *(const v16h*)(wb + 1536);
  const int   n    = (wave << 4) + (lane & 15);
  const float bval = bias[n];
#pragma unroll
  for (int rt = 0; rt < 3; ++rt) {
    const int m0 = rt * 16 + (lane & 15);
    v8f c = {};
    c = WMMA_F16(load_afrag(xa, m0, 0, khalf8), b0, c);
    c = WMMA_F16(load_afrag(xa, m0, 1, khalf8), b1, c);
    c = WMMA_F16(load_afrag(xa, m0, 2, khalf8), b2, c);
    c = WMMA_F16(load_afrag(xa, m0, 3, khalf8), b3, c);
    const int mb = rt * 16 + khalf8;  // C layout: M = vgpr_j + 8*(lane>>4)
#pragma unroll
    for (int j = 0; j < 8; ++j) {
      float  val = c[j] + bval;
      float* dst = out + (mb + j) * H + n;
      if (MODE == 0) *dst = fmaxf(val, 0.f);
      else           *dst += beta * val;
    }
  }
}

// ---------------------------------------------------------------------------
// Fused Q/K/V GEMM: A fragment loaded once from LDS, three WMMAs issued
// against resident B fragments (3 matrices x 4 k-steps held in VGPRs).
// ---------------------------------------------------------------------------
__device__ __forceinline__ void wg_gemm_qkv(
    const _Float16* __restrict__ wq, const _Float16* __restrict__ wk,
    const _Float16* __restrict__ wv, const float* __restrict__ bq,
    const float* __restrict__ bk, const float* __restrict__ bv,
    const _Float16* __restrict__ xa, float* __restrict__ oq,
    float* __restrict__ ok, float* __restrict__ ov) {
  const int wave   = threadIdx.x >> 5;
  const int lane   = threadIdx.x & 31;
  const int khalf8 = (lane >> 4) << 3;
  const _Float16* pq = wq + wave * 2048 + lane * 16;
  const _Float16* pk = wk + wave * 2048 + lane * 16;
  const _Float16* pv = wv + wave * 2048 + lane * 16;
  v16h bqf[4], bkf[4], bvf[4];
#pragma unroll
  for (int ks = 0; ks < 4; ++ks) {
    bqf[ks] = *(const v16h*)(pq + ks * 512);
    bkf[ks] = *(const v16h*)(pk + ks * 512);
    bvf[ks] = *(const v16h*)(pv + ks * 512);
  }
  const int   n   = (wave << 4) + (lane & 15);
  const float bq0 = bq[n], bk0 = bk[n], bv0 = bv[n];
#pragma unroll
  for (int rt = 0; rt < 3; ++rt) {
    const int m0 = rt * 16 + (lane & 15);
    v8f cq = {}, ck = {}, cv = {};
#pragma unroll
    for (int ks = 0; ks < 4; ++ks) {
      const v16h a = load_afrag(xa, m0, ks, khalf8);
      cq = WMMA_F16(a, bqf[ks], cq);
      ck = WMMA_F16(a, bkf[ks], ck);
      cv = WMMA_F16(a, bvf[ks], cv);
    }
    const int mb = rt * 16 + khalf8;
#pragma unroll
    for (int j = 0; j < 8; ++j) {
      const int o = (mb + j) * H + n;
      oq[o] = cq[j] + bq0;
      ok[o] = ck[j] + bk0;
      ov[o] = cv[j] + bv0;
    }
  }
}

// ---------------------------------------------------------------------------
// Fused Dango kernel: one block = 48 rows = 16 trigenic groups.
// ---------------------------------------------------------------------------
__global__ void __launch_bounds__(NTHREADS)
dango_fused(const float* __restrict__ X, const _Float16* __restrict__ wfrag,
            const float* __restrict__ bs, const float* __restrict__ bq,
            const float* __restrict__ bk, const float* __restrict__ bv,
            const float* __restrict__ bo, const float* __restrict__ beta,
            const float* __restrict__ Wp, const float* __restrict__ bp,
            float* __restrict__ out) {
  __shared__ __align__(16) float    s_xdyn[RPB * H];
  __shared__ __align__(16) float    s_stat[RPB * H];
  __shared__ __align__(16) float    s_q[RPB * H];
  __shared__ __align__(16) float    s_k[RPB * H];
  __shared__ __align__(16) float    s_v[RPB * H];
  __shared__ __align__(16) _Float16 s_xa[RPB * H];
  __shared__ float s_rsum[RPB * 4];

  const int t = threadIdx.x;

  // ---- load X tile (48x128 f32) + f16 copy in one pass ---------------
  {
    const float4* src = (const float4*)(X + (size_t)blockIdx.x * (RPB * H));
    float4*       dst = (float4*)s_xdyn;
    v4h*          dsh = (v4h*)s_xa;
    for (int i = t; i < RPB * H / 4; i += NTHREADS) {
      float4 v = src[i];
      dst[i]   = v;
      dsh[i]   = (v4h){(_Float16)v.x, (_Float16)v.y, (_Float16)v.z, (_Float16)v.w};
    }
  }
  __syncthreads();

  // ---- static = relu(X @ Ws + bs) ------------------------------------
  wg_gemm<0>(wfrag, bs, s_xa, s_stat, 0.f);
  __syncthreads();

  // ---- 2 layers of exclude-self MHA + ReZero -------------------------
  for (int l = 0; l < 2; ++l) {
    if (l > 0) {
      const float4* sv = (const float4*)s_xdyn;
      v4h*          dh = (v4h*)s_xa;
      for (int i = t; i < RPB * H / 4; i += NTHREADS) {
        float4 v = sv[i];
        dh[i] = (v4h){(_Float16)v.x, (_Float16)v.y, (_Float16)v.z, (_Float16)v.w};
      }
      __syncthreads();
    }
    const _Float16* wl = wfrag + (size_t)(1 + l * 4) * 16384;
    wg_gemm_qkv(wl, wl + 16384, wl + 2 * 16384, bq + l * H, bk + l * H,
                bv + l * H, s_xa, s_q, s_k, s_v);
    __syncthreads();

    // attention: 48 rows x 4 heads = 192 tasks; softmax over the 2 other
    // group members (self excluded) per head of dim 32.
    if (t < RPB * 4) {
      const int rq = t >> 2;
      const int h  = t & 3;
      const int g  = rq / 3;
      const int lr = rq - g * 3;
      const int k1 = g * 3 + ((lr + 1) % 3);
      const int k2 = g * 3 + ((lr + 2) % 3);
      const float* qr  = s_q + rq * H + h * 32;
      const float* kr1 = s_k + k1 * H + h * 32;
      const float* kr2 = s_k + k2 * H + h * 32;
      float s1 = 0.f, s2 = 0.f;
#pragma unroll
      for (int d = 0; d < 32; ++d) { s1 += qr[d] * kr1[d]; s2 += qr[d] * kr2[d]; }
      const float sc = 0.17677669529663687f;  // 1/sqrt(32)
      s1 *= sc; s2 *= sc;
      const float mx  = fmaxf(s1, s2);
      const float e1  = __expf(s1 - mx), e2 = __expf(s2 - mx);
      const float inv = 1.f / (e1 + e2);
      const float w1 = e1 * inv, w2 = e2 * inv;
      const float* vr1 = s_v + k1 * H + h * 32;
      const float* vr2 = s_v + k2 * H + h * 32;
      v2h* ar = (v2h*)(s_xa + rq * H + h * 32);
#pragma unroll
      for (int d = 0; d < 16; ++d) {
        float a0 = w1 * vr1[2 * d]     + w2 * vr2[2 * d];
        float a1 = w1 * vr1[2 * d + 1] + w2 * vr2[2 * d + 1];
        ar[d] = (v2h){(_Float16)a0, (_Float16)a1};
      }
    }
    __syncthreads();

    // out projection + ReZero residual into running activations
    wg_gemm<2>(wl + 3 * 16384, bo + l * H, s_xa, s_xdyn, beta[l]);
    __syncthreads();
  }

  // ---- gene scores + group mean --------------------------------------
  if (t < RPB * 4) {
    const int r = t >> 2, qd = t & 3;
    float s = 0.f;
    const float* dr = s_xdyn + r * H + qd * 32;
    const float* sr = s_stat + r * H + qd * 32;
    const float* wp = Wp + qd * 32;
#pragma unroll
    for (int c = 0; c < 32; ++c) { float d = dr[c] - sr[c]; s += d * d * wp[c]; }
    s_rsum[t] = s;
  }
  __syncthreads();
  if (t < GPB) {
    float s = 0.f;
#pragma unroll
    for (int j = 0; j < 12; ++j) s += s_rsum[t * 12 + j];
    out[(size_t)blockIdx.x * GPB + t] = s * (1.f / 3.f) + bp[0];
  }
}

// ---------------------------------------------------------------------------
extern "C" void kernel_launch(void* const* d_in, const int* in_sizes, int n_in,
                              void* d_out, int out_size, void* d_ws,
                              size_t ws_size, hipStream_t stream) {
  const float* X    = (const float*)d_in[0];
  // d_in[1] = batch (contiguous repeat(arange) -> not needed)
  const float* Ws   = (const float*)d_in[2];
  const float* bs   = (const float*)d_in[3];
  const float* Wq   = (const float*)d_in[4];
  const float* bq   = (const float*)d_in[5];
  const float* Wk   = (const float*)d_in[6];
  const float* bk   = (const float*)d_in[7];
  const float* Wv   = (const float*)d_in[8];
  const float* bv   = (const float*)d_in[9];
  const float* Wo   = (const float*)d_in[10];
  const float* bo   = (const float*)d_in[11];
  const float* beta = (const float*)d_in[12];
  const float* Wp   = (const float*)d_in[13];
  const float* bp   = (const float*)d_in[14];
  _Float16* wfrag = (_Float16*)d_ws;  // 9*16384 f16 = 288 KB

  repack_weights<<<(9 * 16384 + 255) / 256, 256, 0, stream>>>(Ws, Wq, Wk, Wv,
                                                              Wo, wfrag);

  const int N      = in_sizes[0] / H;  // 393216 rows
  const int blocks = N / RPB;          // 8192 workgroups
  dango_fused<<<blocks, NTHREADS, 0, stream>>>(X, wfrag, bs, bq, bk, bv, bo,
                                               beta, Wp, bp, (float*)d_out);
}
